// TtSwinLayer_70806830841955
// MI455X (gfx1250) — compile-verified, matrix operand order checked by reading, not run on hardware
//
#include <hip/hip_runtime.h>
#include <hip/hip_bf16.h>

typedef __attribute__((ext_vector_type(16))) _Float16 v16h;
typedef __attribute__((ext_vector_type(8)))  float    v8f;
typedef __attribute__((ext_vector_type(4)))  int      v4i;

union HFrag { unsigned u[8]; v16h v; };

#define AS1 __attribute__((address_space(1)))
#define AS3 __attribute__((address_space(3)))

#if defined(__has_builtin)
#if __has_builtin(__builtin_amdgcn_global_load_async_to_lds_b128) && \
    __has_builtin(__builtin_amdgcn_s_wait_asynccnt)
#define USE_ASYNC 1
#endif
#endif
#ifndef USE_ASYNC
#define USE_ASYNC 0
#endif

// K-pair start inside a 16x32 f16 A/B fragment per CDNA5 ISA layout:
// VGPR r (r<4): K = 2r (+8 for hi lanes); VGPR r>=4: K = 16+2(r-4) (+8 hi)
__device__ __forceinline__ int kk_of(int r, int hi) {
    return (r < 4 ? 2 * r : 16 + 2 * (r - 4)) + hi * 8;
}

// ---------------------------------------------------------------------------
// f16 WMMA GEMM: C[M,N] = A[M,K] * BT[N,K]^T (+bias, epilogue variants)
// Block tile 256x128, 256 threads = 8 waves (4x2), wave tile 64x64.
// Double-buffered LDS staging via GLOBAL_LOAD_ASYNC_TO_LDS_B128 (ASYNCcnt).
// EP: 0 = f16 out (+bias)          [QKV]
//     1 = f32 out (+bias)          [attn out proj]
//     2 = f16 out gelu(x+bias)     [MLP up]
//     3 = f32 out x+bias+res       [MLP down + residual]
// ---------------------------------------------------------------------------
template <int EP>
__global__ __launch_bounds__(256) void gemm_wmma(
    const _Float16* __restrict__ A,   // [M,K] row-major
    const _Float16* __restrict__ BT,  // [N,K] row-major (pre-transposed weight)
    const float* __restrict__ bias, const float* __restrict__ res,
    float* __restrict__ outF, _Float16* __restrict__ outH,
    int M, int N, int K)
{
    __shared__ __align__(16) _Float16 As[2][256][40];
    __shared__ __align__(16) _Float16 Bs[2][128][40];

    const int m0 = blockIdx.y * 256;
    const int n0 = blockIdx.x * 128;
    const int tid = threadIdx.x;
    const int lane = tid & 31;
    const int wave = tid >> 5;
    const int wm = wave >> 1;            // 0..3 : 64-row slice
    const int wn = wave & 1;             // 0..1 : 64-col slice
    const int l15 = lane & 15;
    const int hi  = lane >> 4;

    // staging map (per thread): A tile 256x32 halves, B tile 128x32 halves
    const int ar = tid >> 2;             // 0..63 (4 row-steps of 64)
    const int ac = (tid & 3) * 8;        // 0,8,16,24 halves
    const int br = tid >> 1;             // 0..127
    const int bc = (tid & 1) * 16;       // 0,16 halves (+8 second chunk)

    auto stage = [&](int buf, int k0) {
#if USE_ASYNC
        #pragma unroll
        for (int s = 0; s < 4; ++s)
            __builtin_amdgcn_global_load_async_to_lds_b128(
                (AS1 v4i*)&A[(size_t)(m0 + ar + s * 64) * K + k0 + ac],
                (AS3 v4i*)&As[buf][ar + s * 64][ac], 0, 0);
        __builtin_amdgcn_global_load_async_to_lds_b128(
            (AS1 v4i*)&BT[(size_t)(n0 + br) * K + k0 + bc],
            (AS3 v4i*)&Bs[buf][br][bc], 0, 0);
        __builtin_amdgcn_global_load_async_to_lds_b128(
            (AS1 v4i*)&BT[(size_t)(n0 + br) * K + k0 + bc + 8],
            (AS3 v4i*)&Bs[buf][br][bc + 8], 0, 0);
#else
        #pragma unroll
        for (int s = 0; s < 4; ++s)
            *(uint4*)&As[buf][ar + s * 64][ac] =
                *(const uint4*)&A[(size_t)(m0 + ar + s * 64) * K + k0 + ac];
        *(uint4*)&Bs[buf][br][bc] =
            *(const uint4*)&BT[(size_t)(n0 + br) * K + k0 + bc];
        *(uint4*)&Bs[buf][br][bc + 8] =
            *(const uint4*)&BT[(size_t)(n0 + br) * K + k0 + bc + 8];
#endif
    };

    v8f acc[4][4] = {};
    const int NS = K / 32;

    stage(0, 0);
#if USE_ASYNC
    __builtin_amdgcn_s_wait_asynccnt(0);
#endif
    __syncthreads();

    for (int s = 0; s < NS; ++s) {
        int buf = s & 1;
        if (s + 1 < NS) stage(buf ^ 1, (s + 1) * 32);

        HFrag af[4], bf[4];
        #pragma unroll
        for (int mt = 0; mt < 4; ++mt) {
            int row = wm * 64 + mt * 16 + l15;
            #pragma unroll
            for (int r = 0; r < 8; ++r)
                af[mt].u[r] = *(const unsigned*)&As[buf][row][kk_of(r, hi)];
        }
        #pragma unroll
        for (int nt = 0; nt < 4; ++nt) {
            int col = wn * 64 + nt * 16 + l15;
            #pragma unroll
            for (int r = 0; r < 8; ++r)
                bf[nt].u[r] = *(const unsigned*)&Bs[buf][col][kk_of(r, hi)];
        }
        #pragma unroll
        for (int mt = 0; mt < 4; ++mt)
            #pragma unroll
            for (int nt = 0; nt < 4; ++nt)
                acc[mt][nt] = __builtin_amdgcn_wmma_f32_16x16x32_f16(
                    false, af[mt].v, false, bf[nt].v,
                    (short)0, acc[mt][nt], false, false);

#if USE_ASYNC
        __builtin_amdgcn_s_wait_asynccnt(0);
#endif
        __syncthreads();
    }

    // --- epilogue ---
    #pragma unroll
    for (int mt = 0; mt < 4; ++mt) {
        #pragma unroll
        for (int nt = 0; nt < 4; ++nt) {
            int colg = n0 + wn * 64 + nt * 16 + l15;
            float bv = bias[colg];
            #pragma unroll
            for (int r = 0; r < 8; ++r) {
                int rowg = m0 + wm * 64 + mt * 16 + hi * 8 + r;
                size_t o = (size_t)rowg * N + colg;
                float v = acc[mt][nt][r] + bv;
                if (EP == 0) {
                    outH[o] = (_Float16)v;
                } else if (EP == 1) {
                    outF[o] = v;
                } else if (EP == 2) {
                    float g = 0.5f * v * (1.0f + erff(v * 0.70710678118654752f));
                    outH[o] = (_Float16)g;
                } else {
                    outF[o] = v + res[o];
                }
            }
        }
    }
}

// ---------------------------------------------------------------------------
// Weight conversion: f32 -> f16 AND transpose to BT[n][k]; QKV fused [1536,512]
// ---------------------------------------------------------------------------
__global__ __launch_bounds__(256) void convert_weights(
    const float* __restrict__ wq, const float* __restrict__ wk, const float* __restrict__ wv,
    const float* __restrict__ bq, const float* __restrict__ bk, const float* __restrict__ bv,
    const float* __restrict__ wo, const float* __restrict__ wi, const float* __restrict__ wout,
    _Float16* __restrict__ wqkvT, float* __restrict__ bqkv,
    _Float16* __restrict__ woT, _Float16* __restrict__ wiT, _Float16* __restrict__ woutT)
{
    int i = blockIdx.x * 256 + threadIdx.x;     // grid covers 2048*512
    if (i < 1536 * 512) {                       // wqkvT[n][k] : K=512
        int n = i / 512, k = i % 512;
        float v = (n < 512) ? wq[k * 512 + n]
                : (n < 1024) ? wk[k * 512 + n - 512]
                             : wv[k * 512 + n - 1024];
        wqkvT[i] = (_Float16)v;
    }
    if (i < 1536)
        bqkv[i] = (i < 512) ? bq[i] : (i < 1024) ? bk[i - 512] : bv[i - 1024];
    if (i < 512 * 512) {                        // woT[n][k] : K=512
        int n = i / 512, k = i % 512;
        woT[i] = (_Float16)wo[k * 512 + n];
    }
    if (i < 2048 * 512) {
        {   // wiT[n][k] : n<2048, K=512
            int n = i / 512, k = i % 512;
            wiT[i] = (_Float16)wi[k * 2048 + n];
        }
        {   // woutT[n][k] : n<512, K=2048
            int n = i / 2048, k = i % 2048;
            woutT[i] = (_Float16)wout[k * 512 + n];
        }
    }
}

// ---------------------------------------------------------------------------
// LN1 + cyclic shift (-3,-3) + window partition -> f16 win[B*NW*49, 512]
// ---------------------------------------------------------------------------
__global__ __launch_bounds__(256) void ln_shift_win(
    const float* __restrict__ x, const float* __restrict__ g, const float* __restrict__ bta,
    _Float16* __restrict__ win)
{
    __shared__ float red[256];
    int n = blockIdx.x;                  // widx*49 + t
    int widx = n / 49, t = n % 49;
    int b = widx >> 4, w = widx & 15;
    int wy = w >> 2, wx = w & 3;
    int ty = t / 7, tx = t % 7;
    int ys = (wy * 7 + ty + 3) % 28;     // roll(-SHIFT)
    int xs = (wx * 7 + tx + 3) % 28;
    const float* row = x + ((size_t)b * 784 + ys * 28 + xs) * 512;

    int c = threadIdx.x;
    float v0 = row[c], v1 = row[c + 256];
    red[c] = v0 + v1;
    __syncthreads();
    for (int st = 128; st > 0; st >>= 1) { if (c < st) red[c] += red[c + st]; __syncthreads(); }
    float mu = red[0] * (1.0f / 512.0f);
    __syncthreads();
    float d0 = v0 - mu, d1 = v1 - mu;
    red[c] = d0 * d0 + d1 * d1;
    __syncthreads();
    for (int st = 128; st > 0; st >>= 1) { if (c < st) red[c] += red[c + st]; __syncthreads(); }
    float rs = rsqrtf(red[0] * (1.0f / 512.0f) + 1e-5f);

    _Float16* o = win + (size_t)n * 512;
    o[c]       = (_Float16)(d0 * rs * g[c] + bta[c]);
    o[c + 256] = (_Float16)(d1 * rs * g[c + 256] + bta[c + 256]);
}

// ---------------------------------------------------------------------------
// Per (window, head) attention: scores -> softmax -> ctx, all via WMMA.
// 128 threads = 4 waves; padded 64x64 score tile, head-dim K=32.
// ---------------------------------------------------------------------------
__global__ __launch_bounds__(128) void attn_kernel(
    const _Float16* __restrict__ qkv,             // [M,1536] q|k|v
    const float* __restrict__ rel_table,          // [(2*7-1)^2, 16]
    const int* __restrict__ rel_index,            // [49,49]
    _Float16* __restrict__ ctx)                   // [M,512]
{
    __shared__ __align__(16) _Float16 Qs[64][40];
    __shared__ __align__(16) _Float16 Ks[64][40];
    __shared__ __align__(16) _Float16 Vt[32][72];  // Vt[d][j]
    __shared__ float    Ss[64][68];
    __shared__ __align__(16) _Float16 Ps[64][72];

    int widx = blockIdx.x >> 4;
    int h    = blockIdx.x & 15;
    int tid  = threadIdx.x;
    int lane = tid & 31, wave = tid >> 5;
    int l15 = lane & 15, hi = lane >> 4;

    {
        int row = tid >> 1;
        int cc  = (tid & 1) * 16;
        if (row < 49) {
            size_t base = ((size_t)widx * 49 + row) * 1536 + h * 32 + cc;
            *(uint4*)&Qs[row][cc]     = *(const uint4*)&qkv[base];
            *(uint4*)&Qs[row][cc + 8] = *(const uint4*)&qkv[base + 8];
            *(uint4*)&Ks[row][cc]     = *(const uint4*)&qkv[base + 512];
            *(uint4*)&Ks[row][cc + 8] = *(const uint4*)&qkv[base + 520];
            _Float16 tmp[16];
            *(uint4*)&tmp[0] = *(const uint4*)&qkv[base + 1024];
            *(uint4*)&tmp[8] = *(const uint4*)&qkv[base + 1032];
            #pragma unroll
            for (int e = 0; e < 16; ++e) Vt[cc + e][row] = tmp[e];
        } else {
            uint4 z; z.x = z.y = z.z = z.w = 0u;
            *(uint4*)&Qs[row][cc] = z; *(uint4*)&Qs[row][cc + 8] = z;
            *(uint4*)&Ks[row][cc] = z; *(uint4*)&Ks[row][cc + 8] = z;
            #pragma unroll
            for (int e = 0; e < 16; ++e) Vt[cc + e][row] = (_Float16)0.0f;
        }
    }
    __syncthreads();

    // --- scores: S = Q K^T ---
    HFrag a;
    {
        int qrow = wave * 16 + l15;
        #pragma unroll
        for (int r = 0; r < 8; ++r)
            a.u[r] = *(const unsigned*)&Qs[qrow][kk_of(r, hi)];
    }
    const float scale = 0.17677669529663687f;   // 1/sqrt(32)
    int wim = widx & 15;
    int wy = wim >> 2, wx = wim & 3;

    #pragma unroll
    for (int nt = 0; nt < 4; ++nt) {
        HFrag b;
        int col = nt * 16 + l15;
        #pragma unroll
        for (int r = 0; r < 8; ++r)
            b.u[r] = *(const unsigned*)&Ks[col][kk_of(r, hi)];
        v8f s = {};
        s = __builtin_amdgcn_wmma_f32_16x16x32_f16(false, a.v, false, b.v,
                                                   (short)0, s, false, false);
        #pragma unroll
        for (int r = 0; r < 8; ++r) {
            int i = wave * 16 + hi * 8 + r;
            int j = nt * 16 + l15;
            float v;
            if (i < 49 && j < 49) {
                float bias = rel_table[rel_index[i * 49 + j] * 16 + h];
                int tyi = i / 7, txi = i % 7, tyj = j / 7, txj = j % 7;
                int gyi = (wy == 3) ? (tyi < 4 ? 1 : 2) : 0;
                int gxi = (wx == 3) ? (txi < 4 ? 1 : 2) : 0;
                int gyj = (wy == 3) ? (tyj < 4 ? 1 : 2) : 0;
                int gxj = (wx == 3) ? (txj < 4 ? 1 : 2) : 0;
                float msk = ((gyi * 3 + gxi) != (gyj * 3 + gxj)) ? -100.0f : 0.0f;
                v = s[r] * scale + bias + msk;
            } else {
                v = -1e30f;
            }
            Ss[i][j] = v;
        }
    }
    __syncthreads();

    // --- softmax over j (valid 49) ---
    if (tid < 64) {
        float mx = -1e30f;
        for (int j = 0; j < 49; ++j) mx = fmaxf(mx, Ss[tid][j]);
        float sum = 0.0f;
        for (int j = 0; j < 49; ++j) {
            float e = __expf(Ss[tid][j] - mx);
            Ss[tid][j] = e; sum += e;
        }
        float inv = 1.0f / fmaxf(sum, 1e-20f);
        for (int j = 0; j < 49; ++j) Ps[tid][j] = (_Float16)(Ss[tid][j] * inv);
        for (int j = 49; j < 64; ++j) Ps[tid][j] = (_Float16)0.0f;
    }
    __syncthreads();

    // --- ctx = P V : 64x64 (padded) x 64x32 ---
    v8f c2[2] = {};
    #pragma unroll
    for (int ks = 0; ks < 2; ++ks) {
        HFrag a2;
        int prow = wave * 16 + l15;
        #pragma unroll
        for (int r = 0; r < 8; ++r)
            a2.u[r] = *(const unsigned*)&Ps[prow][kk_of(r, hi) + ks * 32];
        #pragma unroll
        for (int nt = 0; nt < 2; ++nt) {
            HFrag b2;
            int d = nt * 16 + l15;
            #pragma unroll
            for (int r = 0; r < 8; ++r)
                b2.u[r] = *(const unsigned*)&Vt[d][kk_of(r, hi) + ks * 32];
            c2[nt] = __builtin_amdgcn_wmma_f32_16x16x32_f16(false, a2.v, false, b2.v,
                                                            (short)0, c2[nt], false, false);
        }
    }
    #pragma unroll
    for (int nt = 0; nt < 2; ++nt) {
        #pragma unroll
        for (int r = 0; r < 8; ++r) {
            int i = wave * 16 + hi * 8 + r;
            int d = nt * 16 + l15;
            if (i < 49)
                ctx[((size_t)widx * 49 + i) * 512 + h * 32 + d] = (_Float16)c2[nt][r];
        }
    }
}

// ---------------------------------------------------------------------------
// window-reverse + unshift + residual -> x1 (f32), then LN2 -> h2 (f16)
// ---------------------------------------------------------------------------
__global__ __launch_bounds__(256) void resid_ln2(
    const float* __restrict__ x, const float* __restrict__ attn,
    const float* __restrict__ g2, const float* __restrict__ b2,
    float* __restrict__ x1, _Float16* __restrict__ h2)
{
    __shared__ float red[256];
    int n = blockIdx.x;
    int b = n / 784, p = n % 784;
    int y = p / 28, xc = p % 28;
    int yp = (y + 25) % 28, xp = (xc + 25) % 28;   // (y - SHIFT) mod 28
    int wy = yp / 7, ty = yp % 7, wx = xp / 7, tx = xp % 7;
    size_t arow = ((size_t)(b * 16 + wy * 4 + wx) * 49 + ty * 7 + tx) * 512;

    int c = threadIdx.x;
    size_t xb = (size_t)n * 512;
    float v0 = x[xb + c]       + attn[arow + c];
    float v1 = x[xb + c + 256] + attn[arow + c + 256];
    x1[xb + c] = v0;
    x1[xb + c + 256] = v1;

    red[c] = v0 + v1;
    __syncthreads();
    for (int st = 128; st > 0; st >>= 1) { if (c < st) red[c] += red[c + st]; __syncthreads(); }
    float mu = red[0] * (1.0f / 512.0f);
    __syncthreads();
    float d0 = v0 - mu, d1 = v1 - mu;
    red[c] = d0 * d0 + d1 * d1;
    __syncthreads();
    for (int st = 128; st > 0; st >>= 1) { if (c < st) red[c] += red[c + st]; __syncthreads(); }
    float rs = rsqrtf(red[0] * (1.0f / 512.0f) + 1e-5f);

    _Float16* o = h2 + xb;
    o[c]       = (_Float16)(d0 * rs * g2[c] + b2[c]);
    o[c + 256] = (_Float16)(d1 * rs * g2[c + 256] + b2[c + 256]);
}

// ---------------------------------------------------------------------------
extern "C" void kernel_launch(void* const* d_in, const int* in_sizes, int n_in,
                              void* d_out, int out_size, void* d_ws, size_t ws_size,
                              hipStream_t stream) {
    const float* x         = (const float*)d_in[0];
    const float* g1        = (const float*)d_in[1];
    const float* b1        = (const float*)d_in[2];
    const float* wq        = (const float*)d_in[3];
    const float* bq        = (const float*)d_in[4];
    const float* wk        = (const float*)d_in[5];
    const float* bk        = (const float*)d_in[6];
    const float* wv        = (const float*)d_in[7];
    const float* bv        = (const float*)d_in[8];
    const float* rel_table = (const float*)d_in[9];
    const float* wo        = (const float*)d_in[10];
    const float* bo        = (const float*)d_in[11];
    const float* g2        = (const float*)d_in[12];
    const float* b2        = (const float*)d_in[13];
    const float* wi        = (const float*)d_in[14];
    const float* bi        = (const float*)d_in[15];
    const float* w_out     = (const float*)d_in[16];
    const float* b_out     = (const float*)d_in[17];
    const int*   rel_index = (const int*)d_in[18];

    const int M = 50176;                      // B * H * W tokens

    char* ws = (char*)d_ws;
    size_t off = 0;
    auto alloc = [&](size_t bytes) -> void* {
        void* p = ws + off;
        off += (bytes + 255) & ~(size_t)255;
        return p;
    };

    _Float16* wqkvT = (_Float16*)alloc((size_t)1536 * 512 * 2);
    float*    bqkv  = (float*)   alloc((size_t)1536 * 4);
    _Float16* woT   = (_Float16*)alloc((size_t)512 * 512 * 2);
    _Float16* wiT   = (_Float16*)alloc((size_t)2048 * 512 * 2);
    _Float16* woutT = (_Float16*)alloc((size_t)512 * 2048 * 2);
    _Float16* win_h = (_Float16*)alloc((size_t)M * 512 * 2);    // reused as h2
    _Float16* qkv_h = (_Float16*)alloc((size_t)M * 1536 * 2);   // reused: attnf, mid
    _Float16* ctx_h = (_Float16*)alloc((size_t)M * 512 * 2);    // contiguous after qkv
    float*    x1    = (float*)   alloc((size_t)M * 512 * 4);

    float*    attnf = (float*)qkv_h;        // qkv dead after attention
    _Float16* mid_h = qkv_h;                // spans qkv+ctx region (== M*2048*2 bytes)
    _Float16* h2    = win_h;                // win dead after QKV GEMM

    // 1) pack/convert/transpose weights to f16
    convert_weights<<<(2048 * 512 + 255) / 256, 256, 0, stream>>>(
        wq, wk, wv, bq, bk, bv, wo, wi, w_out,
        wqkvT, bqkv, woT, wiT, woutT);

    // 2) LN1 + shift + window partition
    ln_shift_win<<<M, 256, 0, stream>>>(x, g1, b1, win_h);

    // 3) fused QKV GEMM: [M,512] x [512,1536] -> f16
    gemm_wmma<0><<<dim3(1536 / 128, M / 256), 256, 0, stream>>>(
        win_h, wqkvT, bqkv, nullptr, nullptr, qkv_h, M, 1536, 512);

    // 4) windowed attention (1024 windows x 16 heads)
    attn_kernel<<<1024 * 16, 128, 0, stream>>>(qkv_h, rel_table, rel_index, ctx_h);

    // 5) output projection: [M,512] x [512,512] -> f32 (+bo)
    gemm_wmma<1><<<dim3(512 / 128, M / 256), 256, 0, stream>>>(
        ctx_h, woT, bo, nullptr, attnf, nullptr, M, 512, 512);

    // 6) window reverse + unshift + residual, LN2
    resid_ln2<<<M, 256, 0, stream>>>(x, attnf, g2, b2, x1, h2);

    // 7) MLP up: [M,512] x [512,2048] -> gelu -> f16
    gemm_wmma<2><<<dim3(2048 / 128, M / 256), 256, 0, stream>>>(
        h2, wiT, bi, nullptr, nullptr, mid_h, M, 2048, 512);

    // 8) MLP down + residual: [M,2048] x [2048,512] + x1 -> d_out (f32)
    gemm_wmma<3><<<dim3(512 / 128, M / 256), 256, 0, stream>>>(
        mid_h, woutT, b_out, x1, (float*)d_out, nullptr, M, 512, 2048);
}
